// ContractiveEquivariantMPlayer_1451698946765
// MI455X (gfx1250) — matrix operand order repaired and checked by reading, not compiled, not run on hardware
//
#include <hip/hip_runtime.h>
#include <hip/hip_bf16.h>
#include <stdint.h>

typedef _Float16 v16h __attribute__((ext_vector_type(16)));
typedef _Float16 v8h  __attribute__((ext_vector_type(8)));
typedef float    v8f  __attribute__((ext_vector_type(8)));

#define FEAT 256
#define NRBF 32
#define CUTOFF 5.0f
#define LDS_STRIDE 264   /* 256 + 8 halves pad: rows land on distinct LDS banks */
#define R_STRIDE   40    /* 32 + 8 halves pad */
#define EDGES_PER_BLOCK 32   /* 2 row-tiles: each B fragment reused 2x */

// ---- WMMA helpers -----------------------------------------------------------

__device__ __forceinline__ v8f wmma_f16(v16h a, v16h b, v8f c) {
  // D = A(16x32 f16) * B(32x16 f16) + C(16x16 f32)
  return __builtin_amdgcn_wmma_f32_16x16x32_f16(false, a, false, b, (short)0, c,
                                                false, false);
}

// A fragment (contiguous K, matches B): lanes 0-15 hold K = 0..15 (+kb),
// lanes 16-31 hold K = 16..31. Caller passes p = &row[kb + rgrp*16].
__device__ __forceinline__ v16h load_a_frag(const _Float16* p) {
  v8h lo = *(const v8h*)(p);        // halves 0..7  -> K = rgrp*16 + 0..7
  v8h hi = *(const v8h*)(p + 8);    // halves 8..15 -> K = rgrp*16 + 8..15
  v16h a;
#pragma unroll
  for (int i = 0; i < 8; ++i) { a[i] = lo[i]; a[i + 8] = hi[i]; }
  return a;
}

// ---- Weight prep: f32 -> f16, transposed to [n][k] --------------------------

__global__ void prep_weights(const float* __restrict__ W1,
                             const float* __restrict__ W2,
                             const float* __restrict__ Wd,
                             _Float16* __restrict__ w1t,
                             _Float16* __restrict__ w2t,
                             _Float16* __restrict__ wdt) {
  int i = blockIdx.x * blockDim.x + threadIdx.x;
  if (i < 256 * 768) {                      // W2: (256,768) row-major
    int k = i / 768, n = i % 768;
    w2t[(size_t)n * 256 + k] = (_Float16)W2[i];
  }
  if (i < 256 * 256) {                      // W1: (256,256)
    int k = i >> 8, n = i & 255;
    w1t[(size_t)n * 256 + k] = (_Float16)W1[i];
  }
  if (i < 32 * 768) {                       // Wd: (32,768)
    int k = i / 768, n = i % 768;
    wdt[(size_t)n * 32 + k] = (_Float16)Wd[i];
  }
}

__global__ void zero_f32(float* __restrict__ a, size_t n) {
  size_t i = (size_t)blockIdx.x * blockDim.x + threadIdx.x;
  if (i < n) a[i] = 0.0f;
}

// ---- Fused message-passing kernel: one block == 32 edges --------------------

__global__ void __launch_bounds__(256)
fused_mp(const float* __restrict__ h,  const float* __restrict__ v,
         const float* __restrict__ d,  const float* __restrict__ u,
         const int*   __restrict__ mapping,
         const float* __restrict__ b1, const float* __restrict__ b2,
         const float* __restrict__ bd,
         const _Float16* __restrict__ w1t, const _Float16* __restrict__ w2t,
         const _Float16* __restrict__ wdt,
         float* __restrict__ dh_acc, float* __restrict__ dv_acc,
         float* __restrict__ cnt) {
  __shared__ alignas(16) _Float16 hA[EDGES_PER_BLOCK * LDS_STRIDE]; // h tile
  __shared__ alignas(16) _Float16 hH[EDGES_PER_BLOCK * LDS_STRIDE]; // silu tile
  __shared__ alignas(16) _Float16 hR[EDGES_PER_BLOCK * R_STRIDE];   // RBF feats
  __shared__ float sEnv[EDGES_PER_BLOCK];
  __shared__ float sU[EDGES_PER_BLOCK][3];
  __shared__ int   sSeg[EDGES_PER_BLOCK];

  const int tid  = threadIdx.x;
  const int lane = tid & 31;
  const int wave = tid >> 5;
  const int rgrp = lane >> 4;   // 0 or 1 (half-wave)
  const int lr   = lane & 15;
  const size_t base = (size_t)blockIdx.x * EDGES_PER_BLOCK;

  // Stage A tile: 32 rows x 256 f32 -> f16 in LDS
#pragma unroll 8
  for (int i = tid; i < EDGES_PER_BLOCK * FEAT; i += 256) {
    int r = i >> 8, c = i & 255;
    hA[r * LDS_STRIDE + c] = (_Float16)h[(base + r) * FEAT + c];
  }

  // Per-edge scalars: RBF features, envelope, unit vector, segment, count
  if (tid < EDGES_PER_BLOCK) {
    size_t e = base + tid;
    float dd = d[e];
    float ed = __expf(-dd);
    const float mu0 = __expf(-CUTOFF);
    const float dmu = (1.0f - mu0) / (float)(NRBF - 1);
    float kk = (2.0f / (float)NRBF) * (1.0f - mu0);
    kk = 1.0f / (kk * kk);
#pragma unroll
    for (int j = 0; j < NRBF; ++j) {
      float t = ed - (mu0 + (float)j * dmu);
      hR[tid * R_STRIDE + j] = (_Float16)__expf(-kk * t * t);
    }
    float env = 0.5f * (__cosf(dd * 3.14159265358979f / CUTOFF) + 1.0f);
    sEnv[tid] = (dd < CUTOFF) ? env : 0.0f;
    sU[tid][0] = u[e * 3 + 0];
    sU[tid][1] = u[e * 3 + 1];
    sU[tid][2] = u[e * 3 + 2];
    int seg = mapping[e];
    sSeg[tid] = seg;
    atomicAdd(&cnt[seg], 1.0f);
  }
  __syncthreads();

  // ---- GEMM1: H = silu(h @ W1 + b1) ----
  // Each wave: 2 of 16 feat-tiles; each B fragment reused for 2 row-tiles.
  for (int t = wave; t < 16; t += 8) {
    const int f0 = t * 16;
    v8f acc[2] = {{}, {}};
#pragma unroll
    for (int kk = 0; kk < 8; ++kk) {
      const int kb = kk * 32;
      v16h b = *(const v16h*)(w1t + (size_t)(f0 + lr) * FEAT + kb + rgrp * 16);
#pragma unroll
      for (int rt = 0; rt < 2; ++rt) {
        v16h a = load_a_frag(&hA[(rt * 16 + lr) * LDS_STRIDE + kb + rgrp * 16]);
        acc[rt] = wmma_f16(a, b, acc[rt]);
      }
    }
    const int f = f0 + lr;
    const float bias = b1[f];
#pragma unroll
    for (int rt = 0; rt < 2; ++rt) {
#pragma unroll
      for (int g = 0; g < 8; ++g) {
        int m = rt * 16 + g + 8 * rgrp;             // C/D layout row
        float x = acc[rt][g] + bias;
        float s = x / (1.0f + __expf(-x));          // silu
        hH[m * LDS_STRIDE + f] = (_Float16)s;
      }
    }
  }
  __syncthreads();

  // ---- GEMM2 (phi = H@W2) + RBF GEMM (e = R@Wd) + combine + scatter ----
  for (int t = wave; t < 16; t += 8) {
    const int f0 = t * 16;

    // Prefetch the v_i cachelines the scatter phase will consume (HBM stream)
#pragma unroll
    for (int rt = 0; rt < 2; ++rt) {
      const size_t e0 = base + rt * 16 + 8 * rgrp;
      __builtin_prefetch(v + ((size_t)e0 * FEAT + f0 + lr) * 3, 0, 1);
    }

    v8f p[2][3] = {{{}, {}, {}}, {{}, {}, {}}};  // phi: [rowtile][branch]
    v8f q[2][3] = {{{}, {}, {}}, {{}, {}, {}}};  // e:   [rowtile][branch]
#pragma unroll
    for (int kk = 0; kk < 8; ++kk) {
      const int kb = kk * 32;
      const _Float16* wb = w2t + (size_t)(f0 + lr) * FEAT + kb + rgrp * 16;
      v16h b0 = *(const v16h*)(wb);
      v16h bA = *(const v16h*)(wb + (size_t)256 * FEAT);
      v16h bB = *(const v16h*)(wb + (size_t)512 * FEAT);
#pragma unroll
      for (int rt = 0; rt < 2; ++rt) {
        v16h a = load_a_frag(&hH[(rt * 16 + lr) * LDS_STRIDE + kb + rgrp * 16]);
        p[rt][0] = wmma_f16(a, b0, p[rt][0]);
        p[rt][1] = wmma_f16(a, bA, p[rt][1]);
        p[rt][2] = wmma_f16(a, bB, p[rt][2]);
      }
    }
    {
      // K = 32 single step for the RBF branch
      const _Float16* wb = wdt + (size_t)(f0 + lr) * NRBF + rgrp * 16;
      v16h b0 = *(const v16h*)(wb);
      v16h bA = *(const v16h*)(wb + (size_t)256 * NRBF);
      v16h bB = *(const v16h*)(wb + (size_t)512 * NRBF);
#pragma unroll
      for (int rt = 0; rt < 2; ++rt) {
        v16h a = load_a_frag(&hR[(rt * 16 + lr) * R_STRIDE + rgrp * 16]);
        q[rt][0] = wmma_f16(a, b0, q[rt][0]);
        q[rt][1] = wmma_f16(a, bA, q[rt][1]);
        q[rt][2] = wmma_f16(a, bB, q[rt][2]);
      }
    }

    const int f = f0 + lr;
    const float c1 = b2[f],       e1b = bd[f];
    const float c2 = b2[256 + f], e2b = bd[256 + f];
    const float c3 = b2[512 + f], e3b = bd[512 + f];
#pragma unroll
    for (int rt = 0; rt < 2; ++rt) {
#pragma unroll
      for (int g = 0; g < 8; ++g) {
        const int m = rt * 16 + g + 8 * rgrp;
        const size_t e = base + m;
        const float env = sEnv[m];
        const float o1 = (p[rt][0][g] + c1) * ((q[rt][0][g] + e1b) * env); // f1
        const float o2 = (p[rt][1][g] + c2) * ((q[rt][1][g] + e2b) * env); // f2
        const float o3 = (p[rt][2][g] + c3) * ((q[rt][2][g] + e3b) * env); // f3
        const int seg = sSeg[m];
        const float* vp = v + ((size_t)e * FEAT + f) * 3;
        const float vx = vp[0], vy = vp[1], vz = vp[2];
        float* dvp = dv_acc + ((size_t)seg * FEAT + f) * 3;
        atomicAdd(dvp + 0, o1 * sU[m][0] + o2 * vx);
        atomicAdd(dvp + 1, o1 * sU[m][1] + o2 * vy);
        atomicAdd(dvp + 2, o1 * sU[m][2] + o2 * vz);
        atomicAdd(dh_acc + (size_t)seg * FEAT + f, o3);
      }
    }
  }
}

// ---- Finalize: divide sums by counts (scatter_mean) -------------------------

__global__ void finalize(float* __restrict__ out, const float* __restrict__ cnt,
                         int ncg) {
  size_t i = (size_t)blockIdx.x * blockDim.x + threadIdx.x;
  size_t nh = (size_t)ncg * FEAT;
  size_t ntot = nh + (size_t)ncg * FEAT * 3;
  if (i >= ntot) return;
  int seg = (i < nh) ? (int)(i >> 8) : (int)((i - nh) / (FEAT * 3));
  out[i] /= fmaxf(cnt[seg], 1.0f);
}

// ---- Launch -----------------------------------------------------------------

extern "C" void kernel_launch(void* const* d_in, const int* in_sizes, int n_in,
                              void* d_out, int out_size, void* d_ws, size_t ws_size,
                              hipStream_t stream) {
  const float* h   = (const float*)d_in[0];
  const float* v   = (const float*)d_in[1];
  const float* d   = (const float*)d_in[2];
  const float* u   = (const float*)d_in[3];
  const int*   map = (const int*)d_in[4];
  const float* W1  = (const float*)d_in[5];
  const float* b1  = (const float*)d_in[6];
  const float* W2  = (const float*)d_in[7];
  const float* b2  = (const float*)d_in[8];
  const float* Wd  = (const float*)d_in[9];
  const float* bd  = (const float*)d_in[10];

  const int N   = in_sizes[2];            // number of edges (200000)
  const int NCG = out_size / (FEAT * 4);  // 256 + 3*256 floats per segment

  // Workspace layout
  char* ws = (char*)d_ws;
  _Float16* w1t = (_Float16*)ws;  ws += (size_t)256 * 256 * sizeof(_Float16);
  _Float16* w2t = (_Float16*)ws;  ws += (size_t)768 * 256 * sizeof(_Float16);
  _Float16* wdt = (_Float16*)ws;  ws += (size_t)768 * 32  * sizeof(_Float16);
  float*    cnt = (float*)ws;

  float* dh_acc = (float*)d_out;                    // (NCG, 256)
  float* dv_acc = dh_acc + (size_t)NCG * FEAT;      // (NCG, 256, 3)

  {
    size_t n = (size_t)out_size;
    zero_f32<<<(unsigned)((n + 255) / 256), 256, 0, stream>>>((float*)d_out, n);
  }
  zero_f32<<<(unsigned)((NCG + 255) / 256), 256, 0, stream>>>(cnt, (size_t)NCG);

  prep_weights<<<(256 * 768 + 255) / 256, 256, 0, stream>>>(W1, W2, Wd, w1t, w2t, wdt);

  fused_mp<<<N / EDGES_PER_BLOCK, 256, 0, stream>>>(h, v, d, u, map, b1, b2, bd,
                                                    w1t, w2t, wdt, dh_acc, dv_acc, cnt);

  {
    size_t ntot = (size_t)NCG * FEAT * 4;
    finalize<<<(unsigned)((ntot + 255) / 256), 256, 0, stream>>>((float*)d_out, cnt, NCG);
  }
}
